// LSTMDecoder_1589137899641
// MI455X (gfx1250) — matrix operand order repaired and checked by reading, not compiled, not run on hardware
//
#include <hip/hip_runtime.h>
#include <hip/hip_bf16.h>
#include <math.h>

#define B_ 16
#define T_ 256
#define D_ 512
#define E_ 512
#define V_ 32000
#define G_ 2048        // 4*D
#define NWG 8
#define SCAN_THREADS 512

#define KSLICE 32                  // K-elements per TDM-staged B slice
#define NSLICES (D_ / KSLICE)      // 16
#define BPAD 8                     // bf16 elements of LDS pad per row (16B)
#define BROW (KSLICE + BPAD)       // 40 elems = 80B LDS row stride (bank-conflict free)

typedef __bf16 bf16_t;
typedef bf16_t v16bf __attribute__((ext_vector_type(16)));
typedef bf16_t v8bf  __attribute__((ext_vector_type(8)));
typedef float  v8f   __attribute__((ext_vector_type(8)));
typedef unsigned int u32x4 __attribute__((ext_vector_type(4)));
typedef int i32x4 __attribute__((ext_vector_type(4)));
typedef int i32x8 __attribute__((ext_vector_type(8)));

// ---------------- WMMA fragment helpers (contiguous k = kbase + hi*16 + j in A and B) ----------------

// A fragment: 16x32 bf16, row-major source A[lda]. lane m = lane&15,
// elem j -> K = kbase + hi*16 + j
__device__ inline v16bf load_a_frag(const bf16_t* A, int lda, int row0, int kbase, int lane) {
  const int m  = lane & 15;
  const int hi = (lane >> 4) & 1;
  const bf16_t* p = A + (size_t)(row0 + m) * lda + kbase + hi * 16;
  v8bf lo  = *(const v8bf*)(p);
  v8bf hi8 = *(const v8bf*)(p + 8);
  return __builtin_shufflevector(lo, hi8, 0,1,2,3,4,5,6,7,8,9,10,11,12,13,14,15);
}

// B fragment: 32x16 bf16 from weight W[N,K] row-major (W supplies B^T):
// lane n = lane&15 ; elem j -> K = kbase + (lane>=16)*16 + j  => one contiguous 32B load.
__device__ inline v16bf load_b_frag(const bf16_t* W, int ldw, int ncol0, int kbase, int lane) {
  const int n  = lane & 15;
  const int hi = (lane >> 4) & 1;
  const bf16_t* p = W + (size_t)(ncol0 + n) * ldw + kbase + hi * 16;
  return *(const v16bf*)(p);
}

// Same B fragment but from an LDS-staged tile with 80B row stride (16B-aligned halves).
__device__ inline v16bf load_b_frag_lds(const bf16_t* bb, int ncol0, int lane) {
  const int n  = lane & 15;
  const int hi = (lane >> 4) & 1;
  const bf16_t* p = bb + (ncol0 + n) * BROW + hi * 16;
  v8bf lo = *(const v8bf*)(p);
  v8bf h8 = *(const v8bf*)(p + 8);
  return __builtin_shufflevector(lo, h8, 0,1,2,3,4,5,6,7,8,9,10,11,12,13,14,15);
}

__device__ inline v8f wmma_bf16(v16bf a, v16bf b, v8f c) {
  return __builtin_amdgcn_wmma_f32_16x16x32_bf16(false, a, false, b, (short)0, c, false, false);
}

__device__ inline float sigf(float x) { return 1.0f / (1.0f + __expf(-x)); }

// ---------------- TDM: stage one [256 x KSLICE] bf16 weight slice into LDS ----------------
// D# per ISA 8.3/8.4: 2D tile, data_size=2B, LDS padding 16B every 64B (row stride 80B).
__device__ inline void tdm_load_b_slice(const bf16_t* gsrc, unsigned lds_off) {
  unsigned long long ga = (unsigned long long)(size_t)gsrc;
  u32x4 g0;
  g0[0] = 1u;                                   // count = 1 valid descriptor
  g0[1] = lds_off;                              // lds_addr (bytes)
  g0[2] = (unsigned)ga;                         // global_addr[31:0]
  g0[3] = (unsigned)(ga >> 32) | (2u << 30);    // global_addr[56:32] | type=2 (image)
  i32x8 g1;
  g1[0] = (1 << 16)                             // data_size = 1 -> 2 bytes
        | (1 << 20)                             // pad_enable
        | (3 << 22)                             // pad_interval: 16 DWORDs (64B)
        | (3 << 25);                            // pad_amount: 4 DWORDs (16B)
  g1[1] = (KSLICE & 0xFFFF) << 16;              // tensor_dim0[15:0] @ bits[63:48]
  g1[2] = (256 & 0xFFFF) << 16;                 // tensor_dim1[15:0] @ bits[95:80]
  g1[3] = (KSLICE & 0xFFFF) << 16;              // tile_dim0 @ bits[127:112]
  g1[4] = 256;                                  // tile_dim1 @ bits[143:128]
  g1[5] = D_;                                   // tensor_dim0_stride[31:0] = 512 elems
  g1[6] = 0;                                    // dim0_stride[47:32]=0, dim1_stride[15:0]=0
  g1[7] = 0;
  i32x4 z4 = {0, 0, 0, 0};
#if __clang_major__ >= 23
  i32x8 z8 = {0, 0, 0, 0, 0, 0, 0, 0};
  __builtin_amdgcn_tensor_load_to_lds(g0, g1, z4, z4, z8, 0);
#else
  __builtin_amdgcn_tensor_load_to_lds(g0, g1, z4, z4, 0);
#endif
}

// ---------------- init: zero barrier counters + h ping-pong buffers ----------------
__global__ void k_init(unsigned* ctrs, bf16_t* hbuf) {
  int i = blockIdx.x * blockDim.x + threadIdx.x;
  if (i < 512) ctrs[i] = 0u;
  if (i < 2 * B_ * 2 * D_) hbuf[i] = (bf16_t)0.0f;   // 2 buffers of [16][1024]
}

// ---------------- embedding gather -> bf16, layout r = t*B + b ----------------
__global__ void k_embed(const int* __restrict__ tok, const float* __restrict__ embed,
                        bf16_t* __restrict__ xb) {
  int r = blockIdx.x;              // 0..4095
  int t = r >> 4, b = r & 15;
  int token = tok[b * T_ + t];
  const float* src = embed + (size_t)token * D_;
  bf16_t* dst = xb + (size_t)r * D_;
  for (int c = threadIdx.x; c < D_; c += blockDim.x)
    dst[c] = (token == 0) ? (bf16_t)0.0f : (bf16_t)src[c];   // padding_idx = 0
}

// ---------------- f32 -> bf16 strided matrix convert ----------------
__global__ void k_cvt(const float* __restrict__ src, int sld, bf16_t* __restrict__ dst,
                      int dld, int rows, int cols) {
  size_t total = (size_t)rows * cols;
  for (size_t i = blockIdx.x * (size_t)blockDim.x + threadIdx.x; i < total;
       i += (size_t)gridDim.x * blockDim.x) {
    size_t r = i / cols, c = i % cols;
    dst[r * dld + c] = (bf16_t)src[r * sld + c];
  }
}

// ---------------- encpre[b][n] = enc[b] @ W_ih0[:, D:].T + b_ih0 + b_hh0 ----------------
__global__ void k_encpre(const float* __restrict__ enc, const float* __restrict__ Wih0,
                         const float* __restrict__ bih0, const float* __restrict__ bhh0,
                         float* __restrict__ encpre) {
  int i = blockIdx.x * blockDim.x + threadIdx.x;
  if (i >= B_ * G_) return;
  int b = i / G_, n = i % G_;
  const float* w = Wih0 + (size_t)n * (D_ + E_) + D_;
  const float* e = enc + (size_t)b * E_;
  float s = bih0[n] + bhh0[n];
  for (int k = 0; k < E_; ++k) s += e[k] * w[k];
  encpre[i] = s;
}

// ---------------- gates_in0[4096,2048] = xb @ Wx^T + encpre[b]  (b == row&15) ----------------
__global__ __launch_bounds__(256) void k_gemm_in0(const bf16_t* __restrict__ xb,
                                                  const bf16_t* __restrict__ Wx,
                                                  const float* __restrict__ encpre,
                                                  float* __restrict__ gin0) {
  const int lane = threadIdx.x & 31;
  const int wv = threadIdx.x >> 5;          // 8 waves
  const int mw = wv >> 2, nw = wv & 3;      // 2 x 4 wave grid
  const int row_base = blockIdx.y * 64 + mw * 32;
  const int col_base = blockIdx.x * 256 + nw * 64;
  v8f acc[2][4] = {};
  for (int k = 0; k < D_; k += 32) {
    v16bf a0 = load_a_frag(xb, D_, row_base,      k, lane);
    v16bf a1 = load_a_frag(xb, D_, row_base + 16, k, lane);
    v16bf b0 = load_b_frag(Wx, D_, col_base,      k, lane);
    v16bf b1 = load_b_frag(Wx, D_, col_base + 16, k, lane);
    v16bf b2 = load_b_frag(Wx, D_, col_base + 32, k, lane);
    v16bf b3 = load_b_frag(Wx, D_, col_base + 48, k, lane);
    acc[0][0] = wmma_bf16(a0, b0, acc[0][0]);
    acc[0][1] = wmma_bf16(a0, b1, acc[0][1]);
    acc[0][2] = wmma_bf16(a0, b2, acc[0][2]);
    acc[0][3] = wmma_bf16(a0, b3, acc[0][3]);
    acc[1][0] = wmma_bf16(a1, b0, acc[1][0]);
    acc[1][1] = wmma_bf16(a1, b1, acc[1][1]);
    acc[1][2] = wmma_bf16(a1, b2, acc[1][2]);
    acc[1][3] = wmma_bf16(a1, b3, acc[1][3]);
  }
  const int hi = (lane >> 4) & 1, nl = lane & 15;
  for (int mi = 0; mi < 2; ++mi)
    for (int ni = 0; ni < 4; ++ni) {
      int col = col_base + ni * 16 + nl;
      for (int r = 0; r < 8; ++r) {
        int row = row_base + mi * 16 + r + 8 * hi;
        gin0[(size_t)row * G_ + col] = acc[mi][ni][r] + encpre[(row & 15) * G_ + col];
      }
    }
}

// ---------------- global split barrier (each index used once per launch) ----------------
__device__ inline void gbar(unsigned* ctrs, int idx) {
  __syncthreads();
  if (threadIdx.x == 0) {
    __threadfence();
    __hip_atomic_fetch_add(&ctrs[idx], 1u, __ATOMIC_RELEASE, __HIP_MEMORY_SCOPE_AGENT);
    while (__hip_atomic_load(&ctrs[idx], __ATOMIC_ACQUIRE, __HIP_MEMORY_SCOPE_AGENT) < NWG)
      __builtin_amdgcn_s_sleep(1);
    __threadfence();
  }
  __syncthreads();
}

// ---------------- pipelined 2-layer LSTM scan, 8 persistent WGs, 1 barrier / phase ----------------
// phase ph: layer0 gates(ph) and layer1 gates(ph-1) both read only prev buffer
// prev = [ h0(ph-1) | h1(ph-2) ]  -> single K=1024 A operand for layer1.
__global__ __launch_bounds__(SCAN_THREADS) void k_scan(
    const float* __restrict__ gin0, const bf16_t* __restrict__ Whh0b,
    const bf16_t* __restrict__ W1cat, const float* __restrict__ bih1,
    const float* __restrict__ bhh1, bf16_t* __restrict__ hbuf,
    bf16_t* __restrict__ outs, unsigned* ctrs) {
  __shared__ float g0s[B_][256];
  __shared__ float g1s[B_][256];
  const int tid = threadIdx.x, lane = tid & 31, wv = tid >> 5, wg = blockIdx.x;
  const int p = wv >> 2, jq = wv & 3;                 // 16 waves: gate-part x j-quarter
  const int nbase = p * D_ + wg * 64 + jq * 16;       // global gate column of this wave's tile
  const int qbase = p * 64 + jq * 16;                 // LDS column
  const int hi = (lane >> 4) & 1, nl = lane & 15;
  const int e0 = tid * 2, e1 = e0 + 1;                // two (b,j) cells per thread
  const int b0 = e0 >> 6, j0 = e0 & 63, b1 = e1 >> 6, j1 = e1 & 63;
  const int jc0 = wg * 64 + j0, jc1 = wg * 64 + j1;
  float c0a = 0.f, c0b = 0.f, c1a = 0.f, c1b = 0.f;   // cell state lives in registers
  const float bias1 = bih1[nbase + nl] + bhh1[nbase + nl];

  for (int ph = 0; ph <= T_; ++ph) {
    bf16_t* prev = hbuf + (size_t)((ph & 1) ^ 1) * (B_ * 2 * D_);
    bf16_t* cur  = hbuf + (size_t)(ph & 1) * (B_ * 2 * D_);
    if (ph < T_) {                                    // layer0 recurrent GEMM, K=512
      v8f acc = {};
      for (int k = 0; k < D_; k += 32) {
        v16bf a = load_a_frag(prev, 2 * D_, 0, k, lane);
        v16bf w = load_b_frag(Whh0b, D_, nbase, k, lane);
        acc = wmma_bf16(a, w, acc);
      }
      const float* gi = gin0 + ((size_t)ph * B_) * G_ + nbase + nl;
      for (int r = 0; r < 8; ++r)
        g0s[r + 8 * hi][qbase + nl] = acc[r] + gi[(size_t)(r + 8 * hi) * G_];
    }
    if (ph >= 1) {                                    // layer1 GEMM, K=1024 over [h0|h1]
      v8f acc = {};
      for (int k = 0; k < 2 * D_; k += 32) {
        v16bf a = load_a_frag(prev, 2 * D_, 0, k, lane);
        v16bf w = load_b_frag(W1cat, 2 * D_, nbase, k, lane);
        acc = wmma_bf16(a, w, acc);
      }
      for (int r = 0; r < 8; ++r)
        g1s[r + 8 * hi][qbase + nl] = acc[r] + bias1;
    }
    __syncthreads();
    if (ph < T_) {                                    // layer0 elementwise -> h0(ph)
      float i_, f_, g_, o_, h_;
      i_ = g0s[b0][j0]; f_ = g0s[b0][64 + j0]; g_ = g0s[b0][128 + j0]; o_ = g0s[b0][192 + j0];
      c0a = sigf(f_) * c0a + sigf(i_) * tanhf(g_); h_ = sigf(o_) * tanhf(c0a);
      cur[b0 * (2 * D_) + jc0] = (bf16_t)h_;
      i_ = g0s[b1][j1]; f_ = g0s[b1][64 + j1]; g_ = g0s[b1][128 + j1]; o_ = g0s[b1][192 + j1];
      c0b = sigf(f_) * c0b + sigf(i_) * tanhf(g_); h_ = sigf(o_) * tanhf(c0b);
      cur[b1 * (2 * D_) + jc1] = (bf16_t)h_;
    }
    if (ph >= 1) {                                    // layer1 elementwise -> h1(ph-1)
      int tq = ph - 1;
      float i_, f_, g_, o_, h_;
      i_ = g1s[b0][j0]; f_ = g1s[b0][64 + j0]; g_ = g1s[b0][128 + j0]; o_ = g1s[b0][192 + j0];
      c1a = sigf(f_) * c1a + sigf(i_) * tanhf(g_); h_ = sigf(o_) * tanhf(c1a);
      cur[b0 * (2 * D_) + D_ + jc0] = (bf16_t)h_;
      outs[((size_t)tq * B_ + b0) * D_ + jc0] = (bf16_t)h_;
      i_ = g1s[b1][j1]; f_ = g1s[b1][64 + j1]; g_ = g1s[b1][128 + j1]; o_ = g1s[b1][192 + j1];
      c1b = sigf(f_) * c1b + sigf(i_) * tanhf(g_); h_ = sigf(o_) * tanhf(c1b);
      cur[b1 * (2 * D_) + D_ + jc1] = (bf16_t)h_;
      outs[((size_t)tq * B_ + b1) * D_ + jc1] = (bf16_t)h_;
    }
    gbar(ctrs, ph);
  }
}

// ---------------- output projection with TDM-staged B tiles ----------------
// C[4096,32000] = outs @ Wo^T + b ; B tile [256 x KSLICE] staged into LDS by the
// Tensor Data Mover (double-buffered, issued by wave 0, s_wait_tensorcnt to sync),
// shared by all 8 waves; row remap scan row t*B+b -> out row b*T+t.
__global__ __launch_bounds__(256) void k_gemm_out(const bf16_t* __restrict__ outs,
                                                  const bf16_t* __restrict__ Wo,
                                                  const float* __restrict__ bout,
                                                  float* __restrict__ out) {
  __shared__ __align__(128) bf16_t lbuf[2][256][BROW];
  const int lane = threadIdx.x & 31;
  const int wv = threadIdx.x >> 5;
  const int mw = wv >> 2, nw = wv & 3;
  const int row_base = blockIdx.y * 64 + mw * 32;
  const int col_base = blockIdx.x * 256;          // workgroup column base
  const int ncl = nw * 64;                        // wave's local column base
  v8f acc[2][4] = {};

  if (threadIdx.x < 32)                           // prologue: stage slice 0
    tdm_load_b_slice(Wo + (size_t)col_base * D_,
                     (unsigned)(size_t)&lbuf[0][0][0]);

  for (int s = 0; s < NSLICES; ++s) {
    if (threadIdx.x < 32) {
      if (s + 1 < NSLICES) {
        tdm_load_b_slice(Wo + (size_t)col_base * D_ + (s + 1) * KSLICE,
                         (unsigned)(size_t)&lbuf[(s + 1) & 1][0][0]);
        __builtin_amdgcn_s_wait_tensorcnt(1);     // TDM in-order: slice s has landed
      } else {
        __builtin_amdgcn_s_wait_tensorcnt(0);
      }
    }
    __syncthreads();                              // LDS slice s visible to all waves
    const bf16_t* bb = &lbuf[s & 1][0][0];
    const int k = s * KSLICE;
    v16bf a0 = load_a_frag(outs, D_, row_base,      k, lane);
    v16bf a1 = load_a_frag(outs, D_, row_base + 16, k, lane);
    v16bf b0 = load_b_frag_lds(bb, ncl,      lane);
    v16bf b1 = load_b_frag_lds(bb, ncl + 16, lane);
    v16bf b2 = load_b_frag_lds(bb, ncl + 32, lane);
    v16bf b3 = load_b_frag_lds(bb, ncl + 48, lane);
    acc[0][0] = wmma_bf16(a0, b0, acc[0][0]);
    acc[0][1] = wmma_bf16(a0, b1, acc[0][1]);
    acc[0][2] = wmma_bf16(a0, b2, acc[0][2]);
    acc[0][3] = wmma_bf16(a0, b3, acc[0][3]);
    acc[1][0] = wmma_bf16(a1, b0, acc[1][0]);
    acc[1][1] = wmma_bf16(a1, b1, acc[1][1]);
    acc[1][2] = wmma_bf16(a1, b2, acc[1][2]);
    acc[1][3] = wmma_bf16(a1, b3, acc[1][3]);
    __syncthreads();                              // slice s consumed before overwrite
  }

  const int hi = (lane >> 4) & 1, nl = lane & 15;
  for (int mi = 0; mi < 2; ++mi)
    for (int ni = 0; ni < 4; ++ni) {
      int col = col_base + ncl + ni * 16 + nl;
      float bias = bout[col];
      for (int r = 0; r < 8; ++r) {
        int rs = row_base + mi * 16 + r + 8 * hi;   // scan row t*16+b
        int orow = (rs & 15) * T_ + (rs >> 4);      // out row b*256+t
        out[(size_t)orow * V_ + col] = acc[mi][ni][r] + bias;
      }
    }
}

// ---------------- launch ----------------
extern "C" void kernel_launch(void* const* d_in, const int* in_sizes, int n_in,
                              void* d_out, int out_size, void* d_ws, size_t ws_size,
                              hipStream_t stream) {
  (void)in_sizes; (void)n_in; (void)out_size; (void)ws_size;
  const int*   tok   = (const int*)  d_in[0];
  const float* enc   = (const float*)d_in[1];
  const float* embed = (const float*)d_in[2];
  const float* Wih0  = (const float*)d_in[3];
  const float* Whh0f = (const float*)d_in[4];
  const float* bih0  = (const float*)d_in[5];
  const float* bhh0  = (const float*)d_in[6];
  const float* Wih1  = (const float*)d_in[7];
  const float* Whh1f = (const float*)d_in[8];
  const float* bih1  = (const float*)d_in[9];
  const float* bhh1  = (const float*)d_in[10];
  const float* Woutf = (const float*)d_in[11];
  const float* bout  = (const float*)d_in[12];
  float* out = (float*)d_out;

  char* ws = (char*)d_ws;
  size_t off = 0;
  auto alloc = [&](size_t bytes) -> char* {
    off = (off + 255) & ~(size_t)255;
    char* ptr = ws + off;
    off += bytes;
    return ptr;
  };
  bf16_t*   xb     = (bf16_t*)alloc((size_t)T_ * B_ * D_ * 2);        // 4 MB
  bf16_t*   Wx     = (bf16_t*)alloc((size_t)G_ * D_ * 2);             // 2 MB
  bf16_t*   Whh0b  = (bf16_t*)alloc((size_t)G_ * D_ * 2);             // 2 MB
  bf16_t*   W1cat  = (bf16_t*)alloc((size_t)G_ * 2 * D_ * 2);         // 4 MB  [W_ih1|W_hh1]
  bf16_t*   Wob    = (bf16_t*)alloc((size_t)V_ * D_ * 2);             // 32.8 MB
  float*    encpre = (float*) alloc((size_t)B_ * G_ * 4);             // 128 KB
  float*    gin0   = (float*) alloc((size_t)T_ * B_ * G_ * 4);        // 33.6 MB
  bf16_t*   hbuf   = (bf16_t*)alloc((size_t)2 * B_ * 2 * D_ * 2);     // 64 KB (ping-pong)
  bf16_t*   outs   = (bf16_t*)alloc((size_t)T_ * B_ * D_ * 2);        // 4 MB
  unsigned* ctrs   = (unsigned*)alloc((size_t)512 * 4);               // barrier counters

  k_init<<<256, 256, 0, stream>>>(ctrs, hbuf);
  k_embed<<<T_ * B_, 256, 0, stream>>>(tok, embed, xb);
  k_cvt<<<512, 256, 0, stream>>>(Wih0, D_ + E_, Wx, D_, G_, D_);         // x-part of W_ih0
  k_cvt<<<512, 256, 0, stream>>>(Whh0f, D_, Whh0b, D_, G_, D_);
  k_cvt<<<512, 256, 0, stream>>>(Wih1, D_, W1cat, 2 * D_, G_, D_);
  k_cvt<<<512, 256, 0, stream>>>(Whh1f, D_, W1cat + D_, 2 * D_, G_, D_);
  k_cvt<<<2048, 256, 0, stream>>>(Woutf, D_, Wob, D_, V_, D_);
  k_encpre<<<(B_ * G_ + 255) / 256, 256, 0, stream>>>(enc, Wih0, bih0, bhh0, encpre);
  k_gemm_in0<<<dim3(G_ / 256, T_ * B_ / 64), 256, 0, stream>>>(xb, Wx, encpre, gin0);
  k_scan<<<NWG, SCAN_THREADS, 0, stream>>>(gin0, Whh0b, W1cat, bih1, bhh1, hbuf, outs, ctrs);
  k_gemm_out<<<dim3(V_ / 256, T_ * B_ / 64), 256, 0, stream>>>(outs, Wob, bout, out);
}
